// NetVLAD_11776800325972
// MI455X (gfx1250) — matrix operand order repaired and checked by reading, not compiled, not run on hardware
//
#include <hip/hip_runtime.h>

// Problem constants (fixed by the reference: N=64, C=512, H=W=30, K=64)
#define NN 64
#define CC 512
#define PP 900
#define KK 64
#define TP 32                          // pixels per tile
#define NT_PIX ((PP + TP - 1) / TP)    // 29 tiles (28 full + 1 of 4 pixels)

typedef __attribute__((ext_vector_type(16))) __bf16 v16bf;
typedef __attribute__((ext_vector_type(8)))  float  v8f;
typedef __attribute__((ext_vector_type(4)))  unsigned int v4u;
typedef __attribute__((ext_vector_type(4)))  int    v4i;
typedef __attribute__((ext_vector_type(8)))  int    v8i;

#if __has_builtin(__builtin_amdgcn_tensor_load_to_lds)
#define HAVE_TDM 1
#if defined(__has_include)
#if __has_include(<hip/amd_detail/amd_gfx1250_TDM.h>)
#define TDM_6ARG 1   // therock-10.0 headers -> 6-arg builtin form
#endif
#endif
#endif

// f32 -> bf16 round-to-nearest-even (u16 storage; avoids scalar __bf16 conversions)
static __device__ inline unsigned short f32_to_bf16(float f) {
    unsigned u = __builtin_bit_cast(unsigned, f);
    unsigned r = 0x7FFFu + ((u >> 16) & 1u);
    return (unsigned short)((u + r) >> 16);
}

// Lane map for WMMA A (16x32) in contiguous k order (not AMD's table order), same as B:
// lanes 0-15: halves 0..15 -> K=0..15
// lanes 16-31: halves 0..15 -> K=16..31
static __device__ inline void amap(int m, int kloc, int& lane, int& j) {
    lane = m + ((kloc >= 16) ? 16 : 0);
    j    = kloc & 15;
}

#ifdef HAVE_TDM
// Issue a 2D TDM tile load (f32 elements) Global -> LDS per CDNA5 D# layout.
// dims/strides in elements; lds_off in bytes.
static __device__ inline void tdm_load_2d_f32(unsigned lds_off, const void* gptr,
                                              unsigned tensor_d0, unsigned tensor_d1,
                                              unsigned tile_d0, unsigned tile_d1,
                                              unsigned stride_d0) {
    unsigned long long ga = (unsigned long long)gptr;
    v4u g0;
    g0[0] = 1u;                                            // count=1, user descriptor
    g0[1] = lds_off;                                       // lds_addr (bytes)
    g0[2] = (unsigned)(ga & 0xFFFFFFFFu);                  // global_addr[31:0]
    g0[3] = (unsigned)((ga >> 32) & 0x01FFFFFFu)           // global_addr[56:32]
          | (2u << 30);                                    // type=2 ("image")
    v8i g1;
    g1[0] = (int)(2u << 16);                               // data_size=2 (4 bytes), wg_mask=0
    g1[1] = (int)((tensor_d0 & 0xFFFFu) << 16);            // tensor_dim0[15:0]
    g1[2] = (int)((tensor_d0 >> 16) & 0xFFFFu)
          | (int)((tensor_d1 & 0xFFFFu) << 16);            // tensor_dim0[31:16] | tensor_dim1[15:0]
    g1[3] = (int)((tensor_d1 >> 16) & 0xFFFFu)
          | (int)((tile_d0 & 0xFFFFu) << 16);              // tensor_dim1[31:16] | tile_dim0
    g1[4] = (int)(tile_d1 & 0xFFFFu);                      // tile_dim1 (tile_dim2=0)
    g1[5] = (int)stride_d0;                                // tensor_dim0_stride[31:0]
    g1[6] = 0;                                             // stride hi / dim1_stride lo
    g1[7] = 0;
    v4i z4 = {0, 0, 0, 0};
#ifdef TDM_6ARG
    v8i z8 = {0, 0, 0, 0, 0, 0, 0, 0};
    __builtin_amdgcn_tensor_load_to_lds(g0, g1, z4, z4, z8, 0);
#else
    __builtin_amdgcn_tensor_load_to_lds(g0, g1, z4, z4, 0);
#endif
}
#endif

struct __align__(64) Smem {
    unsigned short Astage[4][16][32][16]; // conv_w bf16, [m_tile][c_chunk][lane][half]   64 KB
    unsigned short B1[2][16][32][16];     // x_raw (K=chan, N=pix) [pix_tile][c_chunk][lane][half] 32 KB
    unsigned short B2[32][32][16];        // x_raw (K=pix, N=chan) [chan_tile][lane][half] 32 KB
                                          // (B1+B2 = contiguous 64 KB, doubles as TDM scratch at init)
    unsigned short A2[4][32][16];         // a*scale (K=pix) [m_tile][lane][half]          4 KB
    float logits[KK][TP];                 // raw logits, reused in-place as `a`            8 KB
    float scale[TP];
    float red[32][TP];                    // sum-of-squares partials                       4 KB
    float asum[KK];
};

__global__ __launch_bounds__(256)
void netvlad_fused_kernel(const float* __restrict__ x,     // [N][C][P]
                          const float* __restrict__ w,     // [K][C]
                          const float* __restrict__ cent,  // [K][C]
                          float* __restrict__ out) {       // [N][K][C]
    __shared__ Smem s;
    const int n    = blockIdx.x;
    const int tid  = threadIdx.x;
    const int lane = tid & 31;
    const int wave = tid >> 5;
    const float* xb = x + (size_t)n * CC * PP;

    if (tid < KK) s.asum[tid] = 0.f;

    // ---- Stage conv_w as bf16 in WMMA-A layout (once per block) ----
#ifdef HAVE_TDM
    {
        const unsigned scratch_off = (unsigned)(unsigned long long)(&s.B1[0][0][0][0]);
        const float* wsrc = (const float*)(&s.B1[0][0][0][0]);
        for (int ch = 0; ch < 2; ++ch) {           // two 32-cluster chunks of 64 KB
            if (tid < 32) {                        // single wave issues the DMA
                tdm_load_2d_f32(scratch_off, w + (size_t)ch * 32 * CC,
                                /*tensor_d0=*/CC, /*tensor_d1=*/32,
                                /*tile_d0=*/CC,   /*tile_d1=*/32,
                                /*stride_d0=*/CC);
                __builtin_amdgcn_s_wait_tensorcnt(0);
            }
            __syncthreads();                       // LDS tile visible to all waves
            for (int i = tid; i < 32 * CC; i += 256) {
                int k = ch * 32 + (i >> 9);
                int c = i & (CC - 1);
                int l, j; amap(k & 15, c & 31, l, j);
                s.Astage[k >> 4][c >> 5][l][j] = f32_to_bf16(wsrc[i]);
            }
            __syncthreads();                       // scratch reused by next chunk
        }
    }
#else
    for (int i = tid; i < KK * CC; i += 256) {
        int k = i >> 9;
        int c = i & (CC - 1);
        int l, j; amap(k & 15, c & 31, l, j);
        s.Astage[k >> 4][c >> 5][l][j] = f32_to_bf16(w[i]);
    }
    __syncthreads();
#endif

    // ---- Persistent GEMM2 accumulators: this wave owns 16 clusters x 256 channels ----
    const int mt_w   = wave & 3;          // cluster tile (16 clusters)
    const int ntbase = (wave >> 2) * 16;  // channel-tile base (16 tiles of 16 channels)
    v8f acc[16];
    #pragma unroll
    for (int i = 0; i < 16; ++i)
        #pragma unroll
        for (int e = 0; e < 8; ++e) acc[i][e] = 0.f;

    for (int t = 0; t < NT_PIX; ++t) {
        const int pbase = t * TP;
        const int pc    = (PP - pbase < TP) ? (PP - pbase) : TP;

        // prefetch next tile's slice of x into cache
        if (t + 1 < NT_PIX)
            __builtin_prefetch(xb + (size_t)(tid & 127) * PP + pbase + TP, 0, 1);

        // ---- Single pass over x tile: stage RAW x as bf16 (both layouts) + sumsq ----
        // Normalization is deferred: logits get *scale[p]; GEMM2 folds scale into a.
        {
            const int p4   = (tid & 7) * 4;   // this thread's 4-pixel group
            const int crow = tid >> 3;        // channel low 5 bits (0..31)
            float ss[4] = {0.f, 0.f, 0.f, 0.f};
            for (int it = 0; it < 16; ++it) { // 16 channel chunks of 32
                const int c = crow + it * 32;
                const float* src = xb + (size_t)c * PP + pbase + p4;
                float vv[4];
                if (pbase + p4 + 4 <= PP) {
                    float4 v = *(const float4*)src;   // global_load_b128
                    vv[0] = v.x; vv[1] = v.y; vv[2] = v.z; vv[3] = v.w;
                } else {
                    #pragma unroll
                    for (int e = 0; e < 4; ++e)
                        vv[e] = (pbase + p4 + e < PP) ? src[e] : 0.f;
                }
                #pragma unroll
                for (int e = 0; e < 4; ++e) {
                    const int p = p4 + e;
                    const float val = vv[e];
                    ss[e] += val * val;
                    const unsigned short bv = f32_to_bf16(val);
                    // GEMM1 B: K = channel (crow within chunk `it`), N = pixel
                    s.B1[p >> 4][it][(p & 15) + ((crow >= 16) ? 16 : 0)][crow & 15] = bv;
                    // GEMM2 B: K = pixel, N = channel
                    s.B2[c >> 4][(c & 15) + ((p >= 16) ? 16 : 0)][p & 15] = bv;
                }
            }
            #pragma unroll
            for (int e = 0; e < 4; ++e) s.red[crow][p4 + e] = ss[e];
        }
        __syncthreads();
        if (tid < TP) {
            float ssum = 0.f;
            for (int r = 0; r < 32; ++r) ssum += s.red[r][tid];
            s.scale[tid] = (tid < pc) ? (1.f / fmaxf(sqrtf(ssum), 1e-12f)) : 0.f;
        }
        __syncthreads();

        // ---- GEMM1: raw_logits[k,p] = sum_c w[k,c]*x_raw[c,p] (one 16x16 tile/wave) ----
        {
            const int mt = wave & 3;
            const int nt = wave >> 2;   // 0..1 pixel tiles
            v8f accl;
            #pragma unroll
            for (int e = 0; e < 8; ++e) accl[e] = 0.f;
            #pragma unroll
            for (int cchunk = 0; cchunk < 16; ++cchunk) {
                v16bf av = *(const v16bf*)(&s.Astage[mt][cchunk][lane][0]);
                v16bf bv = *(const v16bf*)(&s.B1[nt][cchunk][lane][0]);
                accl = __builtin_amdgcn_wmma_f32_16x16x32_bf16(
                          false, av, false, bv, (short)0, accl, false, false);
            }
            // D layout: M = r + 8*(lane>=16), N = lane&15
            const int p    = nt * 16 + (lane & 15);
            const int krow = mt * 16 + ((lane >> 4) << 3);
            #pragma unroll
            for (int r = 0; r < 8; ++r) s.logits[krow + r][p] = accl[r];
        }
        __syncthreads();

        // ---- Softmax over K per pixel (true logit = raw_logit * scale[p]) ----
        if (tid < TP) {
            const int pl = tid;
            if (pl < pc) {
                const float sc = s.scale[pl];
                float mx = -3.402823466e38f;
                for (int k = 0; k < KK; ++k) mx = fmaxf(mx, s.logits[k][pl] * sc);
                float sum = 0.f;
                for (int k = 0; k < KK; ++k) {
                    float e = __expf(s.logits[k][pl] * sc - mx);
                    s.logits[k][pl] = e;
                    sum += e;
                }
                float inv = 1.f / sum;
                for (int k = 0; k < KK; ++k) s.logits[k][pl] *= inv;
            } else {
                for (int k = 0; k < KK; ++k) s.logits[k][pl] = 0.f;
            }
        }
        __syncthreads();
        if (tid < KK) {   // deterministic asum accumulation (unscaled a)
            float sum = 0.f;
            for (int pl = 0; pl < TP; ++pl) sum += s.logits[tid][pl];
            s.asum[tid] += sum;
        }
        // stage a*scale (bf16, WMMA-A layout, K = pixels) -- folds normalization into GEMM2
        for (int i = tid; i < KK * TP; i += 256) {
            int k = i >> 5, pl = i & 31;
            int l, j; amap(k & 15, pl, l, j);
            s.A2[k >> 4][l][j] = f32_to_bf16(s.logits[k][pl] * s.scale[pl]);
        }
        __syncthreads();

        // ---- GEMM2: vlad[k,c] += sum_p (a*scale)[k,p]*x_raw[p,c] (K=32, one step) ----
        {
            v16bf a2 = *(const v16bf*)(&s.A2[mt_w][lane][0]);
            #pragma unroll
            for (int i = 0; i < 16; ++i) {
                v16bf b2 = *(const v16bf*)(&s.B2[ntbase + i][lane][0]);
                acc[i] = __builtin_amdgcn_wmma_f32_16x16x32_bf16(
                            false, a2, false, b2, (short)0, acc[i], false, false);
            }
        }
        __syncthreads();   // staging buffers reused next tile
    }

    // ---- Epilogue: out[n][k][c] = acc - centroids[k][c] * asum[k] ----
    float* outn = out + (size_t)n * KK * CC;
    const int ncol  = lane & 15;
    const int rbase = (lane >> 4) << 3;
    #pragma unroll
    for (int i = 0; i < 16; ++i) {
        const int c = (ntbase + i) * 16 + ncol;
        #pragma unroll
        for (int r = 0; r < 8; ++r) {
            const int k = mt_w * 16 + rbase + r;
            outn[k * CC + c] = acc[i][r] - cent[k * CC + c] * s.asum[k];
        }
    }
}

extern "C" void kernel_launch(void* const* d_in, const int* in_sizes, int n_in,
                              void* d_out, int out_size, void* d_ws, size_t ws_size,
                              hipStream_t stream) {
    const float* x    = (const float*)d_in[0];  // [64][512][900]
    const float* w    = (const float*)d_in[1];  // [64][512]
    const float* cent = (const float*)d_in[2];  // [64][512]
    float* out        = (float*)d_out;          // [64][64][512]
    (void)in_sizes; (void)n_in; (void)out_size; (void)d_ws; (void)ws_size;
    netvlad_fused_kernel<<<dim3(NN), dim3(256), 0, stream>>>(x, w, cent, out);
}
